// M_3Simp_AB_87273735454747
// MI455X (gfx1250) — compile-verified, matrix operand order checked by reading, not run on hardware
//
#include <hip/hip_runtime.h>
#include <hip/hip_bf16.h>
#include <math.h>

#define BATCH 2
#define SEQ   48
#define DIM   256
#define NH    8
#define HD    32
#define NLM   (SEQ*SEQ)   // 2304
#define NT16  (NLM/16)    // 144 lm tiles of width 16
#define NPAIR (NT16/2)    // 72 K=32 pairs for GEMM2

typedef __attribute__((ext_vector_type(16))) __bf16 v16bf;
typedef __attribute__((ext_vector_type(8)))  float  v8f;

// K index inside a 16x32 bf16 A-fragment (or 32x16 B-fragment, symmetric):
// lanes 0-15 hold K 0-7 (elems 0-7) and K 16-23 (elems 8-15);
// lanes 16-31 hold K 8-15 and K 24-31.  (CDNA5 ISA 7.12.2)
__device__ __forceinline__ int kidx(int j, int half) {
    return j + 8 * half + ((j >= 8) ? 8 : 0);
}

__device__ __forceinline__ float wave_max(float v) {
#pragma unroll
    for (int off = 16; off > 0; off >>= 1)
        v = fmaxf(v, __shfl_xor(v, off, 32));
    return v;
}
__device__ __forceinline__ float wave_sum(float v) {
#pragma unroll
    for (int off = 16; off > 0; off >>= 1)
        v += __shfl_xor(v, off, 32);
    return v;
}

// ---------------------------------------------------------------------------
// Generic 96x256 @ 256x256 + bias GEMM, one 16x16 tile per wave (WMMA bf16).
// mode 0: C = A@W + b        mode 1: C = resid + mish(A@W + b)
// ---------------------------------------------------------------------------
__global__ __launch_bounds__(256)
void gemm_bias_kernel(const float* __restrict__ A, const float* __restrict__ W,
                      const float* __restrict__ bias, const float* __restrict__ resid,
                      float* __restrict__ C, int mode)
{
    const int lane = threadIdx.x & 31;
    const int wave = threadIdx.x >> 5;
    const int half = lane >> 4;
    const int lrow = lane & 15;
    const int tile  = blockIdx.x * 8 + wave;   // 0..95 = 6 mtiles x 16 ntiles
    const int mtile = tile % 6;
    const int ntile = tile / 6;
    const int n = ntile * 16 + lrow;

    v8f acc = {0.f, 0.f, 0.f, 0.f, 0.f, 0.f, 0.f, 0.f};
    for (int kc = 0; kc < 8; ++kc) {
        v16bf a, bm;
#pragma unroll
        for (int j = 0; j < 16; ++j) {
            const int kd = kc * 32 + kidx(j, half);
            a[j]  = (__bf16)A[(mtile * 16 + lrow) * DIM + kd];
            bm[j] = (__bf16)W[kd * DIM + n];
        }
        acc = __builtin_amdgcn_wmma_f32_16x16x32_bf16(false, a, false, bm,
                                                      (short)0, acc, false, false);
    }
    const float bn = bias[n];
#pragma unroll
    for (int r = 0; r < 8; ++r) {
        const int row = mtile * 16 + r + 8 * half;
        float y = acc[r] + bn;
        if (mode == 1) {
            const float x  = y;
            const float sp = (x > 20.f) ? x : log1pf(expf(x));
            y = resid[row * DIM + n] + x * tanhf(sp);
        }
        C[row * DIM + n] = y;
    }
}

// ---------------------------------------------------------------------------
// Build fragment-swizzled pair-product tensors:
//  PTs: B-fragments of P[lm,d]=k2h[l,d]*k3h[m,d] for GEMM1 (L = G x P^T).
//       Record (bh, t, lane) -> 16 bf16 (32 B), consumed as one v16bf load.
//  PVs: B-fragments of PV[lm,d]=v2h[l,d]*v3h[m,d] for GEMM2 (R = E x PV),
//       with the deterministic (t, t+8) tile pairing baked in.
//       Record (bh, pair, nt, lane) -> 16 bf16 (32 B).
// ---------------------------------------------------------------------------
__global__ __launch_bounds__(256)
void build_pv_kernel(const float* __restrict__ k2b, const float* __restrict__ k3b,
                     const float* __restrict__ v2b, const float* __restrict__ v3b,
                     __bf16* __restrict__ PTs, __bf16* __restrict__ PVs)
{
    const int bh = blockIdx.x;           // 0..15
    const int b = bh / NH, h = bh % NH;

    // PTs: 144 tiles x 32 lanes = 4608 records per bh
    for (int slot = threadIdx.x; slot < NT16 * 32; slot += 256) {
        const int t = slot >> 5, lane = slot & 31;
        const int half = lane >> 4, lrow = lane & 15;
        const int lm = t * 16 + lrow;
        const int l = lm / SEQ, m = lm % SEQ;
        v16bf rec;
#pragma unroll
        for (int j = 0; j < 16; ++j) {
            const int cl = h * HD + kidx(j, half);
            rec[j] = (__bf16)(k2b[(b * SEQ + l) * DIM + cl] * k3b[(b * SEQ + m) * DIM + cl]);
        }
        *(v16bf*)(PTs + ((size_t)(bh * NT16 + t) * 32 + lane) * 16) = rec;
    }
    // PVs: 72 pairs x 2 ntiles x 32 lanes = 4608 records per bh
    for (int slot = threadIdx.x; slot < NPAIR * 2 * 32; slot += 256) {
        const int pair = slot >> 6;
        const int nt   = (slot >> 5) & 1;
        const int lane = slot & 31;
        const int half = lane >> 4, lrow = lane & 15;
        const int u = pair >> 3, w = pair & 7;   // pair covers tiles (w+16u, w+16u+8)
        const int d = h * HD + nt * 16 + lrow;
        v16bf rec;
#pragma unroll
        for (int j = 0; j < 16; ++j) {
            const int kl = kidx(j, half);
            const int t  = (kl < 16) ? (w + 16 * u) : (w + 16 * u + 8);
            const int lm = t * 16 + (kl & 15);
            const int l = lm / SEQ, m = lm % SEQ;
            rec[j] = (__bf16)(v2b[(b * SEQ + l) * DIM + d] * v3b[(b * SEQ + m) * DIM + d]);
        }
        *(v16bf*)(PVs + (((size_t)(bh * NPAIR + pair) * 2 + nt) * 32 + lane) * 16) = rec;
    }
}

// ---------------------------------------------------------------------------
// Fused 4-way attention for one (b,h,q), single streaming pass:
//  online (flash-style) softmax per wave over its lm-tiles, WMMA for both
//  L = G x P^T and R += E x PV, end-of-block merge of per-wave (m, s, R).
// ---------------------------------------------------------------------------
__global__ __launch_bounds__(256)
void attn_kernel(const float* __restrict__ qbuf, const float* __restrict__ kbuf,
                 const float* __restrict__ vbuf, const __bf16* __restrict__ PTs,
                 const __bf16* __restrict__ PVs, float* __restrict__ attnout)
{
    const int tid  = threadIdx.x;
    const int lane = tid & 31, wave = tid >> 5;
    const int half = lane >> 4, lrow = lane & 15;
    const int qi = blockIdx.x % SEQ;
    const int bh = blockIdx.x / SEQ;
    const int b  = bh / NH, h = bh % NH;

    // E staged in A-fragment element order: [row][group][j] where the 32 cols
    // split into group 0 = {0-7,16-23} and group 1 = {8-15,24-31}.
    __shared__ __align__(32) __bf16 Elds[8][SEQ][2][16];   // 24 KB
    __shared__ float  Rlds[SEQ][HD];                       // 6 KB
    __shared__ float  redm[8], reds[8];
    __shared__ float  bmax_s, bsum_s;

    for (int i = tid; i < SEQ * HD; i += 256) (&Rlds[0][0])[i] = 0.f;
    __syncthreads();

    const float scale = 0.17677669529663687f; // 1/sqrt(HD)
    const float* qrow = qbuf + (size_t)(b * SEQ + qi) * DIM + h * HD;

    // A-fragments of G[k,d] = scale * qh[q,d] * kh[k,d]  (48x32 -> 3 frags)
    v16bf g[3];
#pragma unroll
    for (int mt = 0; mt < 3; ++mt) {
#pragma unroll
        for (int j = 0; j < 16; ++j) {
            const int d  = kidx(j, half);
            const int kk = mt * 16 + lrow;
            g[mt][j] = (__bf16)(scale * qrow[d] * kbuf[(b * SEQ + kk) * DIM + h * HD + d]);
        }
    }

    const __bf16* PTb = PTs + (size_t)bh * NT16 * 32 * 16;
    const __bf16* PVb = PVs + (size_t)bh * NPAIR * 2 * 32 * 16;

    // E-write target decomposition for col = piece*16 + lrow:
    //   group g = (col>>3)&1,  elem j = (col&7) + 8*(col>>4)
    const int eg[2] = { (lrow >> 3) & 1, ((16 + lrow) >> 3) & 1 };
    const int ej[2] = { (lrow & 7),      (lrow & 7) + 8 };

    v8f racc[3][2];
#pragma unroll
    for (int mt = 0; mt < 3; ++mt)
#pragma unroll
        for (int nt = 0; nt < 2; ++nt)
            racc[mt][nt] = (v8f){0.f, 0.f, 0.f, 0.f, 0.f, 0.f, 0.f, 0.f};
    float lsum = 0.f;
    float m    = -3.4e38f;   // wave-local running max (uniform across lanes)

    for (int u = 0; u < 9; ++u) {
        const int tt0 = wave + 16 * u;   // this wave's tiles (tt0, tt0+8)

        // --- logits for the whole K=32 chunk (both 16-wide pieces) ---
        v8f c[2][3];
#pragma unroll
        for (int piece = 0; piece < 2; ++piece) {
            const int t = tt0 + piece * 8;
            const v16bf bm = *(const v16bf*)(PTb + ((size_t)t * 32 + lane) * 16);
#pragma unroll
            for (int mt = 0; mt < 3; ++mt) {
                v8f z = {0.f, 0.f, 0.f, 0.f, 0.f, 0.f, 0.f, 0.f};
                c[piece][mt] = __builtin_amdgcn_wmma_f32_16x16x32_bf16(
                    false, g[mt], false, bm, (short)0, z, false, false);
            }
        }

        // --- online max update + rescale of running state ---
        float cmax = -3.4e38f;
#pragma unroll
        for (int piece = 0; piece < 2; ++piece)
#pragma unroll
            for (int mt = 0; mt < 3; ++mt)
#pragma unroll
                for (int r = 0; r < 8; ++r)
                    cmax = fmaxf(cmax, c[piece][mt][r]);
        cmax = wave_max(cmax);
        const float mnew = fmaxf(m, cmax);
        const float f = __expf(m - mnew);   // wave-uniform; exp(-inf)=0 first time
        m = mnew;
        lsum *= f;
#pragma unroll
        for (int mt = 0; mt < 3; ++mt)
#pragma unroll
            for (int nt = 0; nt < 2; ++nt)
#pragma unroll
                for (int r = 0; r < 8; ++r)
                    racc[mt][nt][r] *= f;

        // --- exponentiate and stage E in A-fragment order ---
#pragma unroll
        for (int piece = 0; piece < 2; ++piece)
#pragma unroll
            for (int mt = 0; mt < 3; ++mt)
#pragma unroll
                for (int r = 0; r < 8; ++r) {
                    const float e = __expf(c[piece][mt][r] - m);   // <= 1
                    lsum += e;
                    Elds[wave][mt * 16 + r + 8 * half][eg[piece]][ej[piece]] = (__bf16)e;
                }

        // --- R += E x PV for this K=32 chunk ---
        const int pair = u * 8 + wave;
        v16bf pb[2];
#pragma unroll
        for (int nt = 0; nt < 2; ++nt)
            pb[nt] = *(const v16bf*)(PVb + (((size_t)pair * 2 + nt) * 32 + lane) * 16);
#pragma unroll
        for (int mt = 0; mt < 3; ++mt) {
            const v16bf ae = *(const v16bf*)&Elds[wave][mt * 16 + lrow][half][0];
#pragma unroll
            for (int nt = 0; nt < 2; ++nt)
                racc[mt][nt] = __builtin_amdgcn_wmma_f32_16x16x32_bf16(
                    false, ae, false, pb[nt], (short)0, racc[mt][nt], false, false);
        }
    }

    // --- merge per-wave (m, s, R) across the block ---
    lsum = wave_sum(lsum);
    if (lane == 0) { redm[wave] = m; reds[wave] = lsum; }
    __syncthreads();
    if (tid == 0) {
        float M = redm[0];
        for (int w = 1; w < 8; ++w) M = fmaxf(M, redm[w]);
        float S = 0.f;
        for (int w = 0; w < 8; ++w) S += reds[w] * __expf(redm[w] - M);
        bmax_s = M;
        bsum_s = S;
    }
    __syncthreads();
    const float fw = __expf(m - bmax_s);   // wave-uniform merge factor
#pragma unroll
    for (int mt = 0; mt < 3; ++mt)
#pragma unroll
        for (int nt = 0; nt < 2; ++nt)
#pragma unroll
            for (int r = 0; r < 8; ++r)
                atomicAdd(&Rlds[mt * 16 + r + 8 * half][nt * 16 + lrow],
                          racc[mt][nt][r] * fw);
    __syncthreads();

    if (tid < HD) {
        const int d = tid;
        float acc = 0.f;
        for (int kk = 0; kk < SEQ; ++kk)
            acc += vbuf[(b * SEQ + kk) * DIM + h * HD + d] * Rlds[kk][d];
        attnout[(size_t)(b * SEQ + qi) * DIM + h * HD + d] = acc / bsum_s;
    }
}

// ---------------------------------------------------------------------------
extern "C" void kernel_launch(void* const* d_in, const int* in_sizes, int n_in,
                              void* d_out, int out_size, void* d_ws, size_t ws_size,
                              hipStream_t stream)
{
    const float* Q   = (const float*)d_in[0];
    const float* K   = (const float*)d_in[1];
    const float* Wq  = (const float*)d_in[2];  const float* bq  = (const float*)d_in[3];
    const float* Wk  = (const float*)d_in[4];  const float* bk  = (const float*)d_in[5];
    const float* Wv  = (const float*)d_in[6];  const float* bv  = (const float*)d_in[7];
    const float* Wk2 = (const float*)d_in[8];  const float* bk2 = (const float*)d_in[9];
    const float* Wv2 = (const float*)d_in[10]; const float* bv2 = (const float*)d_in[11];
    const float* Wk3 = (const float*)d_in[12]; const float* bk3 = (const float*)d_in[13];
    const float* Wv3 = (const float*)d_in[14]; const float* bv3 = (const float*)d_in[15];
    const float* Wo  = (const float*)d_in[16]; const float* bo  = (const float*)d_in[17];

    char* ws = (char*)d_ws;
    const size_t PROJ = (size_t)BATCH * SEQ * DIM * sizeof(float);  // 96 KB each
    float* qb  = (float*)(ws + 0 * PROJ);
    float* kb  = (float*)(ws + 1 * PROJ);
    float* vb  = (float*)(ws + 2 * PROJ);
    float* v2b = (float*)(ws + 3 * PROJ);
    float* k2b = (float*)(ws + 4 * PROJ);
    float* v3b = (float*)(ws + 5 * PROJ);
    float* k3b = (float*)(ws + 6 * PROJ);
    float* ab  = (float*)(ws + 7 * PROJ);
    __bf16* PTs = (__bf16*)(ws + 8 * PROJ);
    __bf16* PVs = (__bf16*)(ws + 8 * PROJ +
                            (size_t)BATCH * NH * NT16 * 32 * 16 * sizeof(__bf16));

    const dim3 gthr(256);
    // stage 1 projections (q, k, v from inputs)
    gemm_bias_kernel<<<dim3(12), gthr, 0, stream>>>(Q,  Wq,  bq,  nullptr, qb,  0);
    gemm_bias_kernel<<<dim3(12), gthr, 0, stream>>>(K,  Wk,  bk,  nullptr, kb,  0);
    gemm_bias_kernel<<<dim3(12), gthr, 0, stream>>>(K,  Wv,  bv,  nullptr, vb,  0);
    // stage 2 projections (from projected k — faithful to source)
    gemm_bias_kernel<<<dim3(12), gthr, 0, stream>>>(kb, Wv2, bv2, nullptr, v2b, 0);
    gemm_bias_kernel<<<dim3(12), gthr, 0, stream>>>(kb, Wk2, bk2, nullptr, k2b, 0);
    gemm_bias_kernel<<<dim3(12), gthr, 0, stream>>>(kb, Wv3, bv3, nullptr, v3b, 0);
    gemm_bias_kernel<<<dim3(12), gthr, 0, stream>>>(kb, Wk3, bk3, nullptr, k3b, 0);
    // fragment-swizzled pair-product tensors (L2 resident: 4.7 MB)
    build_pv_kernel<<<dim3(BATCH * NH), gthr, 0, stream>>>(k2b, k3b, v2b, v3b, PTs, PVs);
    // fused 4-way attention, one block per (b,h,q), single streaming pass
    attn_kernel<<<dim3(BATCH * NH * SEQ), gthr, 0, stream>>>(qb, kb, vb, PTs, PVs, ab);
    // final: out = attn + mish(attn @ Wo + bo)
    gemm_bias_kernel<<<dim3(12), gthr, 0, stream>>>(ab, Wo, bo, ab, (float*)d_out, 1);
}
